// NeuralSDEGenerator_7584912244982
// MI455X (gfx1250) — matrix operand order repaired and checked by reading, not compiled, not run on hardware
//
#include <hip/hip_runtime.h>
#include <math.h>

// ---- problem constants (match reference) ----
#define B_TOT   16384
#define T_STEPS 10
#define M_DIM   8
#define RDIM    256
#define DIN     16
#define HDIM    32
#define NT      32          // batch columns per workgroup (2 N-tiles of 16)

typedef __attribute__((ext_vector_type(16))) __bf16 v16bf;
typedef __attribute__((ext_vector_type(8)))  __bf16 v8bf;
typedef __attribute__((ext_vector_type(8)))  float  v8f;

// Native CDNA5 transcendental tanh (TRANS32). Prefer the clang builtin if this
// toolchain declares it; otherwise emit v_tanh_f32 directly (v_nop satisfies
// the 1-independent-op TRANS hazard rule from the ISA).
#if defined(__has_builtin) && __has_builtin(__builtin_amdgcn_tanhf)
__device__ __forceinline__ float fast_tanh(float x) { return __builtin_amdgcn_tanhf(x); }
#elif defined(__has_builtin) && __has_builtin(__builtin_amdgcn_tanh_f32)
__device__ __forceinline__ float fast_tanh(float x) { return __builtin_amdgcn_tanh_f32(x); }
#else
__device__ __forceinline__ float fast_tanh(float x) {
    float r;
    asm volatile("v_tanh_f32 %0, %1\n\tv_nop" : "=v"(r) : "v"(x));
    return r;
}
#endif

__device__ __forceinline__ unsigned short f2bf(float x) {
    union { float f; unsigned u; } c; c.f = x;
    unsigned u = c.u;
    unsigned r = u + 0x7FFFu + ((u >> 16) & 1u);   // round-to-nearest-even
    return (unsigned short)(r >> 16);
}

// Prep: convert B1 (64K floats) + B2 (512K floats) to bf16 matrices in d_ws.
__global__ void cvt_mats_bf16(const float* __restrict__ B1,
                              const float* __restrict__ B2,
                              unsigned short* __restrict__ ws, int n) {
    int i = blockIdx.x * blockDim.x + threadIdx.x;
    if (i >= n) return;
    float v = (i < RDIM * RDIM) ? B1[i] : B2[i - RDIM * RDIM];
    ws[i] = f2bf(v);
}

__global__ __launch_bounds__(256)
void sde_wmma_kernel(const float* __restrict__ V,
                     const float* __restrict__ inc,
                     const float* __restrict__ W1,
                     const float* __restrict__ b1,
                     const float* __restrict__ W2,
                     const float* __restrict__ b2,
                     const float* __restrict__ rho1p,
                     const float* __restrict__ rho2p,
                     const float* __restrict__ rho3p,
                     const float* __restrict__ rho4p,
                     const float* __restrict__ lam1,
                     const float* __restrict__ lam2,
                     const float* __restrict__ Wro,
                     const float* __restrict__ bro,
                     const unsigned short* __restrict__ mats,
                     float* __restrict__ out)
{
    __shared__ float wroLds[T_STEPS * RDIM];                       // 10 KB
    __shared__ float c1Lds[RDIM];                                  // rho2*lam1
    __shared__ float c2Lds[M_DIM * RDIM];                          // rho4*lam2
    __shared__ float dwLds[(T_STEPS - 1) * M_DIM * NT];            // dW tile (9 KB)
    __shared__ float hLds[NT * HDIM];                              // init MLP hidden (4 KB)
    __shared__ unsigned short RbLds[RDIM * NT] __attribute__((aligned(32))); // bf16 state: [kchunk][col32][32] (16 KB)
    __shared__ float outacc[T_STEPS * NT];                         // 1.25 KB

    const int tid  = threadIdx.x;
    const int w    = tid >> 5;        // wave 0..7 -> owns state rows 32w..32w+31
    const int lane = tid & 31;
    const int col  = lane & 15;       // N index within a 16-wide N-tile
    const int kh   = lane >> 4;       // hi half-wave bit
    const int col0 = blockIdx.x * NT;

    const float rho1 = rho1p[0], rho2 = rho2p[0], rho3 = rho3p[0], rho4 = rho4p[0];

    // ---------------- preload ----------------
    for (int i = tid; i < T_STEPS * RDIM; i += 256) wroLds[i] = Wro[i];
    for (int i = tid; i < RDIM;           i += 256) c1Lds[i]  = rho2 * lam1[i];
    for (int i = tid; i < M_DIM * RDIM;   i += 256) c2Lds[i]  = rho4 * lam2[i];

    // dW: t=0 -> inc[:,0]+inc[:,1];  t>=1 -> inc[:,t+1]   (cumsum with W[0] forced to 0)
    for (int v = tid; v < (T_STEPS - 1) * M_DIM * NT; v += 256) {
        int t   = v / (M_DIM * NT);
        int rem = v - t * (M_DIM * NT);
        int m   = rem >> 5;
        int c   = rem & 31;
        int b   = col0 + c;
        float d;
        if (t == 0) d = inc[b * (T_STEPS * M_DIM) + 0 * M_DIM + m]
                      + inc[b * (T_STEPS * M_DIM) + 1 * M_DIM + m];
        else        d = inc[b * (T_STEPS * M_DIM) + (t + 1) * M_DIM + m];
        dwLds[v] = d;
    }

    // init MLP hidden: h[c][j] = tanh(b1[j] + V[b]·W1[j])   (32 cols x 32 j / 256 thr = 4 each)
    {
        int c  = tid & 31;
        int jq = tid >> 5;
        int b  = col0 + c;
        for (int j = 4 * jq; j < 4 * jq + 4; ++j) {
            float s = b1[j];
            #pragma unroll
            for (int d = 0; d < DIN; ++d) s += V[b * DIN + d] * W1[j * DIN + d];
            hLds[c * HDIM + j] = fast_tanh(s);
        }
    }
    for (int i = tid; i < T_STEPS * NT; i += 256) outacc[i] = bro[i >> 5];
    __syncthreads();

    // ---------------- initial state R0, kept in registers (WMMA C/D layout) ----------------
    const int r0base = 32 * w + kh * 8;        // rows r0base+g (g=0..7) and r0base+16+g
    float rs[2][2][8];                          // [ntile][mhalf][g]
    float c1a[8], c1b[8];
    #pragma unroll
    for (int g = 0; g < 8; ++g) {
        int ra = r0base + g, rb_ = r0base + 16 + g;
        c1a[g] = c1Lds[ra]; c1b[g] = c1Lds[rb_];
        #pragma unroll
        for (int n = 0; n < 2; ++n) {
            int cL = n * 16 + col;
            float sa = b2[ra], sb = b2[rb_];
            for (int j = 0; j < HDIM; ++j) {
                float hv = hLds[cL * HDIM + j];
                sa += W2[ra  * HDIM + j] * hv;
                sb += W2[rb_ * HDIM + j] * hv;
            }
            rs[n][0][g] = sa; rs[n][1][g] = sb;
        }
    }

    auto writeRb = [&]() {   // pack fp32 state -> bf16 B-operand layout (wave owns K-chunk w)
        unsigned* p = (unsigned*)RbLds;
        #pragma unroll
        for (int n = 0; n < 2; ++n) {
            #pragma unroll
            for (int g = 0; g < 8; g += 2) {
                unsigned lo0 = (unsigned)f2bf(rs[n][0][g]) | ((unsigned)f2bf(rs[n][0][g + 1]) << 16);
                unsigned lo1 = (unsigned)f2bf(rs[n][1][g]) | ((unsigned)f2bf(rs[n][1][g + 1]) << 16);
                int e0 = (w * NT + n * 16 + col) * 32 + kh * 16 + g;  // ushort index (even)
                p[e0 >> 1]        = lo0;
                p[(e0 + 8) >> 1]  = lo1;
            }
        }
    };
    auto readout = [&](int ts) {  // out[b][ts] partial: dot(state column, Wro[ts])
        #pragma unroll
        for (int n = 0; n < 2; ++n) {
            float s = 0.f;
            #pragma unroll
            for (int g = 0; g < 8; ++g) {
                s += rs[n][0][g] * wroLds[ts * RDIM + r0base + g];
                s += rs[n][1][g] * wroLds[ts * RDIM + r0base + 16 + g];
            }
            s += __shfl_xor(s, 16);
            if (kh == 0) atomicAdd(&outacc[ts * NT + n * 16 + col], s);
        }
    };

    writeRb();
    readout(0);
    __syncthreads();

    // ---------------- SDE scan: 9 steps x 9 matrices, bf16 WMMA GEMMs ----------------
    const int rowA0 = 32 * w + col;        // A-fragment row (M-tile 32w..32w+15)
    const int rowA1 = rowA0 + 16;

    #pragma unroll 1
    for (int t = 0; t < T_STEPS - 1; ++t) {
        #pragma unroll 1
        for (int i = 0; i < 1 + M_DIM; ++i) {
            const unsigned short* Am = mats + (size_t)i * RDIM * RDIM;
            if (i < M_DIM) __builtin_prefetch(mats + (size_t)(i + 1) * RDIM * RDIM + rowA0 * RDIM, 0, 0);

            v8f acc[2][2];   // [ntile][mhalf]
            #pragma unroll
            for (int n = 0; n < 2; ++n)
                #pragma unroll
                for (int h = 0; h < 2; ++h)
                    acc[n][h] = (v8f){0.f,0.f,0.f,0.f,0.f,0.f,0.f,0.f};

            #pragma unroll
            for (int kc = 0; kc < 8; ++kc) {
                const int ke = kc * 32 + kh * 8;
                union { v16bf v; v8bf h[2]; } a0, a1;
                a0.h[0] = *(const v8bf*)(Am + rowA0 * RDIM + ke);
                a0.h[1] = *(const v8bf*)(Am + rowA0 * RDIM + ke + 16);
                a1.h[0] = *(const v8bf*)(Am + rowA1 * RDIM + ke);
                a1.h[1] = *(const v8bf*)(Am + rowA1 * RDIM + ke + 16);
                const v16bf bf0 = *(const v16bf*)(RbLds + (kc * NT +      col) * 32 + kh * 16);
                const v16bf bf1 = *(const v16bf*)(RbLds + (kc * NT + 16 + col) * 32 + kh * 16);
                acc[0][0] = __builtin_amdgcn_wmma_f32_16x16x32_bf16(false, a0.v, false, bf0,
                                                                    (short)0, acc[0][0], false, false);
                acc[0][1] = __builtin_amdgcn_wmma_f32_16x16x32_bf16(false, a1.v, false, bf0,
                                                                    (short)0, acc[0][1], false, false);
                acc[1][0] = __builtin_amdgcn_wmma_f32_16x16x32_bf16(false, a0.v, false, bf1,
                                                                    (short)0, acc[1][0], false, false);
                acc[1][1] = __builtin_amdgcn_wmma_f32_16x16x32_bf16(false, a1.v, false, bf1,
                                                                    (short)0, acc[1][1], false, false);
            }
            if (i == 0) {          // drift = tanh(rho1*B1@R + rho2*lam1)
                #pragma unroll
                for (int n = 0; n < 2; ++n)
                    #pragma unroll
                    for (int g = 0; g < 8; ++g) {
                        rs[n][0][g] += fast_tanh(rho1 * acc[n][0][g] + c1a[g]);
                        rs[n][1][g] += fast_tanh(rho1 * acc[n][1][g] + c1b[g]);
                    }
            } else {               // diff_m = tanh(rho3*B2[m]@R + rho4*lam2[m]) * dW[m]
                const int m = i - 1;
                #pragma unroll
                for (int n = 0; n < 2; ++n) {
                    const float dwv = dwLds[(t * M_DIM + m) * NT + n * 16 + col];
                    #pragma unroll
                    for (int g = 0; g < 8; ++g) {
                        rs[n][0][g] += fast_tanh(rho3 * acc[n][0][g] + c2Lds[m * RDIM + r0base + g]) * dwv;
                        rs[n][1][g] += fast_tanh(rho3 * acc[n][1][g] + c2Lds[m * RDIM + r0base + 16 + g]) * dwv;
                    }
                }
            }
        }
        __syncthreads();          // everyone done reading old Rb
        writeRb();                // publish new state as bf16 B operand
        readout(t + 1);
        __syncthreads();          // new Rb + outacc visible
    }

    for (int i = tid; i < T_STEPS * NT; i += 256) {
        int tt = i >> 5, cc = i & 31;
        out[(size_t)(col0 + cc) * T_STEPS + tt] = outacc[i];
    }
}

extern "C" void kernel_launch(void* const* d_in, const int* in_sizes, int n_in,
                              void* d_out, int out_size, void* d_ws, size_t ws_size,
                              hipStream_t stream) {
    (void)in_sizes; (void)n_in; (void)out_size; (void)ws_size;
    const float* V    = (const float*)d_in[0];
    const float* inc  = (const float*)d_in[1];
    const float* W1   = (const float*)d_in[2];
    const float* b1   = (const float*)d_in[3];
    const float* W2   = (const float*)d_in[4];
    const float* b2   = (const float*)d_in[5];
    const float* rho1 = (const float*)d_in[6];
    const float* rho2 = (const float*)d_in[7];
    const float* rho3 = (const float*)d_in[8];
    const float* rho4 = (const float*)d_in[9];
    const float* B1   = (const float*)d_in[10];
    const float* B2   = (const float*)d_in[11];
    const float* lam1 = (const float*)d_in[12];
    const float* lam2 = (const float*)d_in[13];
    const float* Wro  = (const float*)d_in[14];
    const float* bro  = (const float*)d_in[15];
    float* out = (float*)d_out;
    unsigned short* mats = (unsigned short*)d_ws;   // 9 * 256*256 bf16 = 1.125 MB

    const int nmat = (1 + M_DIM) * RDIM * RDIM;
    cvt_mats_bf16<<<(nmat + 255) / 256, 256, 0, stream>>>(B1, B2, mats, nmat);

    sde_wmma_kernel<<<B_TOT / NT, 256, 0, stream>>>(
        V, inc, W1, b1, W2, b2, rho1, rho2, rho3, rho4,
        lam1, lam2, Wro, bro, mats, out);
}
